// LSTMModel_5403068858580
// MI455X (gfx1250) — compile-verified, matrix operand order checked by reading, not run on hardware
//
#include <hip/hip_runtime.h>
#include <hip/hip_bf16.h>

// ---------------------------------------------------------------------------
// LSTM decoder for MI455X (gfx1250, wave32, WMMA).
// Strategy: bf16 WMMA f32-accumulate for all GEMMs; weights converted to bf16
// once per call into workspace. Per LSTM layer-step, each wave owns one
// (m_tile, n_tile) 16x16 block of H and computes the i/f/g/o gate fragments
// (N columns n, n+512, n+1024, n+1536 of the [256,2048] gate matrix) so the
// cell update runs fully in registers. h is double-buffered per step parity.
// Projection is fused through 32KB LDS per workgroup.
// ---------------------------------------------------------------------------

typedef __attribute__((ext_vector_type(16))) __bf16 bf16x16;
typedef __attribute__((ext_vector_type(8)))  float  f32x8;

static constexpr int kH   = 512;    // hidden
static constexpr int kB   = 256;    // batch
static constexpr int kT   = 200;    // steps
static constexpr int kL   = 3;      // layers
static constexpr int kG   = 2048;   // 4*H
static constexpr int kH2  = 1024;   // 2*H
static constexpr int kV   = 100;    // vocab
static constexpr int kVp  = 112;    // vocab padded to 7 N-tiles

static __device__ __forceinline__ unsigned short f32_to_bf16(float f) {
  unsigned int u = __float_as_uint(f);
  unsigned int r = (u + 0x7FFFu + ((u >> 16) & 1u)) >> 16;
  return (unsigned short)r;
}

static __device__ __forceinline__ float sigf(float x) {
  return 1.0f / (1.0f + expf(-x));
}

// Load a 16x32 bf16 A/B fragment slice for this lane.
// base already includes: row*stride + half*8 (half = lane/16).
// elements 0..7  <- K offsets {0..7},  elements 8..15 <- K offsets {16..23}.
static __device__ __forceinline__ bf16x16 load_frag(const unsigned short* base) {
  union { bf16x16 v; uint4 u[2]; } f;
  f.u[0] = *(const uint4*)(base);
  f.u[1] = *(const uint4*)(base + 16);
  return f.v;
}

static __device__ __forceinline__ f32x8 wmma_bf16(bf16x16 a, bf16x16 b, f32x8 c) {
  return __builtin_amdgcn_wmma_f32_16x16x32_bf16(
      /*neg_a=*/false, a, /*neg_b=*/false, b,
      /*c_mod=*/(short)0, c, /*reuse_a=*/false, /*reuse_b=*/false);
}

// ------------------------------ prep kernels -------------------------------

__global__ void k_f32_to_bf16(const float* __restrict__ s,
                              unsigned short* __restrict__ d, int n) {
  int i = blockIdx.x * blockDim.x + threadIdx.x;
  if (i < n) d[i] = f32_to_bf16(s[i]);
}

__global__ void k_pad_w2(const float* __restrict__ s,          // [100,1024]
                         unsigned short* __restrict__ d) {     // [112,1024]
  int i = blockIdx.x * blockDim.x + threadIdx.x;
  if (i < kVp * kH2) {
    int row = i >> 10, col = i & (kH2 - 1);
    d[i] = (row < kV) ? f32_to_bf16(s[row * kH2 + col]) : (unsigned short)0;
  }
}

__global__ void k_xemb(const float* __restrict__ embed, const int* __restrict__ sidx,
                       unsigned short* __restrict__ d) {       // [256,512]
  int i = blockIdx.x * blockDim.x + threadIdx.x;
  if (i < kB * kH) {
    int col = i & (kH - 1);
    d[i] = f32_to_bf16(embed[sidx[0] * kH + col]);
  }
}

__global__ void k_zero_f32(float* __restrict__ p, int n) {
  int i = blockIdx.x * blockDim.x + threadIdx.x;
  if (i < n) p[i] = 0.0f;
}

// -------------------- visual projection GEMM (bf16 out) --------------------
// y = x @ w.T + bias   (x:[256,512] bf16, w:[512,512] bf16, y bf16)
// 512 waves, one 16x16 tile each. grid <<<256, 64>>>.
__global__ __launch_bounds__(64) void k_gemm_visual(
    const unsigned short* __restrict__ x, const unsigned short* __restrict__ w,
    const float* __restrict__ bias, unsigned short* __restrict__ y) {
  int lane = threadIdx.x & 31;
  int wave = (blockIdx.x * blockDim.x + threadIdx.x) >> 5;  // 0..511
  int m_tile = wave >> 5, n_tile = wave & 31;
  int row16 = lane & 15, half = lane >> 4;

  f32x8 acc = {0, 0, 0, 0, 0, 0, 0, 0};
  const unsigned short* arow = x + (m_tile * 16 + row16) * kH + half * 8;
  const unsigned short* brow = w + (n_tile * 16 + row16) * kH + half * 8;
#pragma unroll 4
  for (int kb = 0; kb < kH / 32; ++kb) {
    bf16x16 a = load_frag(arow + kb * 32);
    bf16x16 b = load_frag(brow + kb * 32);
    acc = wmma_bf16(a, b, acc);
  }
  int n = n_tile * 16 + row16;
  float bv = bias[n];
#pragma unroll
  for (int r = 0; r < 8; ++r) {
    int m = m_tile * 16 + r + 8 * half;
    y[m * kH + n] = f32_to_bf16(acc[r] + bv);
  }
}

// ------------------------------ LSTM step ----------------------------------
// One layer, one time step. Each wave: (m_tile 0..15, n_tile 0..31), gates
// i/f/g/o accumulated as 4 fragments over K = [x|h] (2*512).
// grid <<<256, 64>>> = 512 waves.
__global__ __launch_bounds__(64) void k_lstm_step(
    const unsigned short* __restrict__ x,    // [256,512] bf16
    const unsigned short* __restrict__ hin,  // [256,512] bf16
    float* __restrict__ c,                   // [256,512] f32 (in/out)
    unsigned short* __restrict__ hout,       // [256,512] bf16
    const unsigned short* __restrict__ wih,  // [2048,512] bf16
    const unsigned short* __restrict__ whh,  // [2048,512] bf16
    const float* __restrict__ bih,           // [2048]
    const float* __restrict__ bhh,           // [2048]
    unsigned short* __restrict__ res,        // [B,T,H] bf16 (top layer) or null
    int t) {
  int lane = threadIdx.x & 31;
  int wave = (blockIdx.x * blockDim.x + threadIdx.x) >> 5;  // 0..511
  int m_tile = wave >> 5, n_tile = wave & 31;
  int row16 = lane & 15, half = lane >> 4;

  f32x8 acc[4];
#pragma unroll
  for (int g = 0; g < 4; ++g) acc[g] = (f32x8){0, 0, 0, 0, 0, 0, 0, 0};

  const unsigned short* arow_x = x + (m_tile * 16 + row16) * kH + half * 8;
  const unsigned short* arow_h = hin + (m_tile * 16 + row16) * kH + half * 8;
  int colW = n_tile * 16 + row16;  // column within H (and within each gate)

#pragma unroll 2
  for (int kb = 0; kb < kH / 32; ++kb) {
    bf16x16 a = load_frag(arow_x + kb * 32);
#pragma unroll
    for (int g = 0; g < 4; ++g) {
      bf16x16 b = load_frag(wih + (g * kH + colW) * kH + kb * 32 + half * 8);
      acc[g] = wmma_bf16(a, b, acc[g]);
    }
  }
#pragma unroll 2
  for (int kb = 0; kb < kH / 32; ++kb) {
    bf16x16 a = load_frag(arow_h + kb * 32);
#pragma unroll
    for (int g = 0; g < 4; ++g) {
      bf16x16 b = load_frag(whh + (g * kH + colW) * kH + kb * 32 + half * 8);
      acc[g] = wmma_bf16(a, b, acc[g]);
    }
  }

  float bsum[4];
#pragma unroll
  for (int g = 0; g < 4; ++g) bsum[g] = bih[g * kH + colW] + bhh[g * kH + colW];

#pragma unroll
  for (int r = 0; r < 8; ++r) {
    int m = m_tile * 16 + r + 8 * half;
    int idx = m * kH + colW;
    float gi = sigf(acc[0][r] + bsum[0]);
    float gf = sigf(acc[1][r] + bsum[1]);
    float gg = tanhf(acc[2][r] + bsum[2]);
    float go = sigf(acc[3][r] + bsum[3]);
    float cn = gf * c[idx] + gi * gg;
    float hn = go * tanhf(cn);
    c[idx] = cn;
    unsigned short hb = f32_to_bf16(hn);
    hout[idx] = hb;
    if (res) res[(m * kT + t) * kH + colW] = hb;  // [B,T,H]
  }
}

// --------------------------- fused projection ------------------------------
// Per 16-row block of res[B*T, H]:
//   stage1: hmid[16,1024] = gelu(res @ w1.T + b1) -> LDS (bf16)
//   stage2: logits[16,112] = hmid @ w2.T + b2 -> scattered to out[B,V,T] (f32)
// grid <<<3200, 256>>> (8 waves / WG).
__global__ __launch_bounds__(256) void k_proj(
    const unsigned short* __restrict__ res,  // [51200, 512] bf16
    const unsigned short* __restrict__ w1,   // [1024, 512] bf16
    const float* __restrict__ b1,            // [1024]
    const unsigned short* __restrict__ w2,   // [112, 1024] bf16 (zero-padded)
    const float* __restrict__ b2,            // [100]
    float* __restrict__ out) {               // [256, 100, 200]
  __shared__ unsigned short lds[16 * kH2];   // 32 KB
  int lane = threadIdx.x & 31;
  int wave = threadIdx.x >> 5;               // 0..7
  int row16 = lane & 15, half = lane >> 4;
  int rbase = blockIdx.x * 16;               // global row base (b*T + t)

  // ---- stage 1: hmid tiles; wave handles 8 of 64 N-tiles ----
  f32x8 acc[8];
#pragma unroll
  for (int j = 0; j < 8; ++j) acc[j] = (f32x8){0, 0, 0, 0, 0, 0, 0, 0};

  const unsigned short* arow = res + (rbase + row16) * kH + half * 8;
  for (int kb = 0; kb < kH / 32; ++kb) {
    bf16x16 a = load_frag(arow + kb * 32);
#pragma unroll
    for (int j = 0; j < 8; ++j) {
      int n = wave * 8 + j;  // hmid N-tile 0..63
      bf16x16 b = load_frag(w1 + (n * 16 + row16) * kH + kb * 32 + half * 8);
      acc[j] = wmma_bf16(a, b, acc[j]);
    }
  }
#pragma unroll
  for (int j = 0; j < 8; ++j) {
    int col = (wave * 8 + j) * 16 + row16;
    float bv = b1[col];
#pragma unroll
    for (int r = 0; r < 8; ++r) {
      float v = acc[j][r] + bv;
      v = 0.5f * v * (1.0f + erff(v * 0.70710678118654752f));  // exact GELU
      lds[(r + 8 * half) * kH2 + col] = f32_to_bf16(v);
    }
  }
  __syncthreads();

  // ---- stage 2: 7 output tiles (V padded to 112); waves 0..6 ----
  if (wave < 7) {
    f32x8 o = {0, 0, 0, 0, 0, 0, 0, 0};
    const unsigned short* brow = w2 + (wave * 16 + row16) * kH2 + half * 8;
    for (int kb = 0; kb < kH2 / 32; ++kb) {
      const unsigned short* ab = &lds[row16 * kH2 + kb * 32 + half * 8];
      union { bf16x16 v; uint4 u[2]; } fa;
      fa.u[0] = *(const uint4*)(ab);
      fa.u[1] = *(const uint4*)(ab + 16);
      bf16x16 b = load_frag(brow + kb * 32);
      o = wmma_bf16(fa.v, b, o);
    }
    int v = wave * 16 + row16;
    float bv = (v < kV) ? b2[v] : 0.0f;
#pragma unroll
    for (int r = 0; r < 8; ++r) {
      int grow = rbase + r + 8 * half;     // = b*T + t
      int bb = grow / kT, tt = grow - bb * kT;
      if (v < kV) out[(bb * kV + v) * kT + tt] = o[r] + bv;
    }
  }
}

// ------------------------------- launcher ----------------------------------

extern "C" void kernel_launch(void* const* d_in, const int* in_sizes, int n_in,
                              void* d_out, int out_size, void* d_ws, size_t ws_size,
                              hipStream_t stream) {
  (void)in_sizes; (void)n_in; (void)out_size; (void)ws_size;

  const float* feat      = (const float*)d_in[0];
  const float* visual_w  = (const float*)d_in[1];
  const float* visual_b  = (const float*)d_in[2];
  const float* embed     = (const float*)d_in[3];
  const float* w_ih      = (const float*)d_in[4];
  const float* w_hh      = (const float*)d_in[5];
  const float* b_ih      = (const float*)d_in[6];
  const float* b_hh      = (const float*)d_in[7];
  const float* proj_w1   = (const float*)d_in[8];
  const float* proj_b1   = (const float*)d_in[9];
  const float* proj_w2   = (const float*)d_in[10];
  const float* proj_b2   = (const float*)d_in[11];
  const int*   start_idx = (const int*)d_in[12];
  float* out = (float*)d_out;

  // workspace bump allocator (~71 MB total; res_bf dominates at 52 MB)
  char* p = (char*)d_ws;
  auto alloc = [&](size_t bytes) -> char* {
    char* r = p;
    p += (bytes + 255) & ~(size_t)255;
    return r;
  };
  unsigned short* wih_bf  = (unsigned short*)alloc((size_t)kL * kG * kH * 2);
  unsigned short* whh_bf  = (unsigned short*)alloc((size_t)kL * kG * kH * 2);
  unsigned short* vw_bf   = (unsigned short*)alloc((size_t)kH * kH * 2);
  unsigned short* pw1_bf  = (unsigned short*)alloc((size_t)kH2 * kH * 2);
  unsigned short* pw2_bf  = (unsigned short*)alloc((size_t)kVp * kH2 * 2);
  unsigned short* feat_bf = (unsigned short*)alloc((size_t)kB * kH * 2);
  unsigned short* feat1_bf= (unsigned short*)alloc((size_t)kB * kH * 2);
  unsigned short* h0_bf   = (unsigned short*)alloc((size_t)kB * kH * 2);
  unsigned short* xemb_bf = (unsigned short*)alloc((size_t)kB * kH * 2);
  unsigned short* hbuf    = (unsigned short*)alloc((size_t)kL * 2 * kB * kH * 2);
  float*          cbuf    = (float*)alloc((size_t)kL * kB * kH * 4);
  unsigned short* res_bf  = (unsigned short*)alloc((size_t)kB * kT * kH * 2);

  auto Hb = [&](int l, int parity) { return hbuf + ((size_t)l * 2 + parity) * kB * kH; };
  auto cvt = [&](const float* s, unsigned short* d, int n) {
    k_f32_to_bf16<<<(n + 255) / 256, 256, 0, stream>>>(s, d, n);
  };

  // ---- weight / input conversion ----
  cvt(w_ih, wih_bf, kL * kG * kH);
  cvt(w_hh, whh_bf, kL * kG * kH);
  cvt(visual_w, vw_bf, kH * kH);
  cvt(proj_w1, pw1_bf, kH2 * kH);
  cvt(feat, feat_bf, kB * kH);
  k_pad_w2<<<(kVp * kH2 + 255) / 256, 256, 0, stream>>>(proj_w2, pw2_bf);
  k_xemb<<<(kB * kH + 255) / 256, 256, 0, stream>>>(embed, start_idx, xemb_bf);
  k_zero_f32<<<(kL * kB * kH + 255) / 256, 256, 0, stream>>>(cbuf, kL * kB * kH);

  // ---- visual projection: feat1 = vp(feat); h0 = vp(feat1) ----
  k_gemm_visual<<<256, 64, 0, stream>>>(feat_bf, vw_bf, visual_b, feat1_bf);
  k_gemm_visual<<<256, 64, 0, stream>>>(feat1_bf, vw_bf, visual_b, h0_bf);

  // ---- 200 decode steps x 3 layers (h ping-pong by step parity) ----
  for (int t = 0; t < kT; ++t) {
    int par = t & 1;
    for (int l = 0; l < kL; ++l) {
      const unsigned short* xin =
          (l == 0) ? ((t == 0) ? xemb_bf : Hb(2, par)) : Hb(l - 1, 1 - par);
      const unsigned short* hin = (t == 0) ? h0_bf : Hb(l, par);
      unsigned short* ho = Hb(l, 1 - par);
      k_lstm_step<<<256, 64, 0, stream>>>(
          xin, hin, cbuf + (size_t)l * kB * kH, ho,
          wih_bf + (size_t)l * kG * kH, whh_bf + (size_t)l * kG * kH,
          b_ih + (size_t)l * kG, b_hh + (size_t)l * kG,
          (l == kL - 1) ? res_bf : (unsigned short*)nullptr, t);
    }
  }

  // ---- fused projection + transpose to [B, V, T] ----
  k_proj<<<(kB * kT) / 16, 256, 0, stream>>>(res_bf, pw1_bf, proj_b1, pw2_bf,
                                             proj_b2, out);
}
